// Bi_attention_42666205119022
// MI455X (gfx1250) — matrix operand order repaired and checked by reading, not compiled, not run on hardware
//
#include <hip/hip_runtime.h>
#include <hip/hip_bf16.h>

#define DIMD 64
#define NTOK 4096
#define NB   4

typedef __attribute__((ext_vector_type(16))) _Float16     v16h;
typedef __attribute__((ext_vector_type(8)))  float        v8f;
typedef __attribute__((ext_vector_type(4)))  unsigned int u32x4;
typedef __attribute__((ext_vector_type(8)))  int          i32x8;
typedef __attribute__((ext_vector_type(4)))  int          i32x4;

union HOp { v16h h; u32x4 q[2]; _Float16 e[16]; };

// A-matrix (16x32 f16) layout: lane holds row (lane&15); low lanes: K = koff+0..15,
// high lanes: K = koff+16..31. Two contiguous 16B loads.
__device__ __forceinline__ HOp load_A(const _Float16* base, int rstride, int row, int koff, int lane) {
  int lo = koff + ((lane & 16) ? 16 : 0);
  const _Float16* p = base + row * rstride;
  HOp o;
  o.q[0] = *(const u32x4*)(p + lo);
  o.q[1] = *(const u32x4*)(p + lo + 8);
  return o;
}
// B-matrix (32x16 f16) layout: lane holds col; low lanes K = koff+0..15, high lanes K = koff+16..31.
// "row" here indexes the memory row whose contiguous dim is K (i.e. q2[m][d] with K=d).
__device__ __forceinline__ HOp load_B(const _Float16* base, int rstride, int row, int koff, int lane) {
  int lo = koff + ((lane & 16) ? 16 : 0);
  const _Float16* p = base + row * rstride;
  HOp o;
  o.q[0] = *(const u32x4*)(p + lo);
  o.q[1] = *(const u32x4*)(p + lo + 8);
  return o;
}
// B-matrix where K runs along memory rows (strided): gather 16 halves per lane from LDS tile.
__device__ __forceinline__ HOp load_BT(const _Float16* base, int rstride, int col, int lane) {
  int klo = (lane & 16) ? 16 : 0;
  HOp o;
#pragma unroll
  for (int i = 0; i < 16; ++i) o.e[i] = base[(klo + i) * rstride + col];
  return o;
}
__device__ __forceinline__ v8f wmma16(const HOp& a, const HOp& b, v8f c) {
  return __builtin_amdgcn_wmma_f32_16x16x32_f16(false, a.h, false, b.h, (short)0, c, false, false);
}

// TDM: async-load a 32-row x 64-half tile (row-major, stride 64 halves) from global into LDS.
// Descriptor per CDNA5 ISA. 6-arg builtin (clang-23 / therock-10.0 lane).
__device__ __forceinline__ void tdm_load_tile(const _Float16* gsrc, _Float16* ldsdst) {
  unsigned long long ga = (unsigned long long)(uintptr_t)gsrc;
  u32x4 g0;
  g0[0] = 1u;                                           // count=1, is_restore=0
  g0[1] = (unsigned int)(uintptr_t)ldsdst;              // lds_addr (low 32 bits = LDS offset)
  g0[2] = (unsigned int)ga;                             // global_addr[31:0]  -> bits 95:64
  g0[3] = (unsigned int)((ga >> 32) & 0x01FFFFFFull) | (2u << 30);  // addr[56:32] | type=2
  i32x8 g1;
  g1[0] = 1 << 16;                 // data_size = 2 bytes
  g1[1] = (int)(64u << 16);        // tensor_dim0 = 64       (bits 79:48)
  g1[2] = (int)(4096u << 16);      // tensor_dim1 = 4096     (bits 111:80)
  g1[3] = (int)(64u << 16);        // tile_dim0 = 64         (bits 127:112)
  g1[4] = 32;                      // tile_dim1 = 32         (bits 143:128)
  g1[5] = 64;                      // tensor_dim0_stride = 64 (bits 207:160)
  g1[6] = (int)(4096u << 16);      // tensor_dim1_stride low bits (255:208)
  g1[7] = 0;
  i32x4 g2 = {0, 0, 0, 0};         // 2D tensor: groups 2/3 unused
  i32x4 g3 = {0, 0, 0, 0};
  i32x8 g4 = {0, 0, 0, 0, 0, 0, 0, 0};
  __builtin_amdgcn_tensor_load_to_lds(g0, g1, g2, g3, g4, 0);
  __builtin_amdgcn_s_wait_tensorcnt(0);
}

// ---------------- projection kernels: q = in @ W^T + bias, stored as f16 ----------------
__global__ __launch_bounds__(256) void prep_q1(const float* __restrict__ x,
                                               const float* __restrict__ w1,
                                               const float* __restrict__ b1,
                                               _Float16* __restrict__ q1h) {
  int tid = blockIdx.x * 256 + threadIdx.x;           // B*N*D = 1048576 threads
  int d = tid & 63;
  int row = tid >> 6;                                 // b*N + n
  const float* xr = x + (size_t)row * DIMD;
  const float* wr = w1 + d * DIMD;
  float acc = b1[d];
#pragma unroll
  for (int k = 0; k < DIMD; ++k) acc += xr[k] * wr[k];
  q1h[tid] = (_Float16)acc;
}

__global__ __launch_bounds__(256) void prep_q2(const float* __restrict__ y,
                                               const float* __restrict__ w2,
                                               const float* __restrict__ b2,
                                               _Float16* __restrict__ q2h) {
  int tid = blockIdx.x * 256 + threadIdx.x;
  int d = tid & 63;
  int bm = tid >> 6;
  int b = bm >> 12;
  int m = bm & 4095;                                  // m = h*W + w
  const float* yc = y + (size_t)b * DIMD * NTOK + m;  // y[b][k][m], stride NTOK over k
  const float* wr = w2 + d * DIMD;
  float acc = b2[d];
#pragma unroll
  for (int k = 0; k < DIMD; ++k) acc += yc[(size_t)k * NTOK] * wr[k];
  q2h[tid] = (_Float16)acc;
}

// ---------------- Pass A: flash attention -> v2, row stats (max, sumexp) ----------------
__global__ __launch_bounds__(256) void attn_passA(const _Float16* __restrict__ q1h,
                                                  const _Float16* __restrict__ q2h,
                                                  float* __restrict__ v2,
                                                  float* __restrict__ Mst,
                                                  float* __restrict__ Lst) {
  __shared__ alignas(16) _Float16 sq2[2][32 * DIMD];
  __shared__ alignas(16) _Float16 sp[8][16 * 32];
  const int tid = threadIdx.x, lane = tid & 31, wave = tid >> 5;
  const int col = lane & 15, rb = (lane & 16) ? 8 : 0;
  const int b = blockIdx.x >> 5;
  const int nrow0 = ((blockIdx.x & 31) << 7) + wave * 16;

  const _Float16* q1base = q1h + (size_t)(b * NTOK + nrow0) * DIMD;
  HOp a0 = load_A(q1base, DIMD, col, 0, lane);
  HOp a1 = load_A(q1base, DIMD, col, 32, lane);

  HOp onesB;                                 // all-ones B tile: P @ 1 gives row sums on the matrix pipe
#pragma unroll
  for (int i = 0; i < 16; ++i) onesB.e[i] = (_Float16)1.0f;

  v8f zeroV = {};
  v8f o[4];
#pragma unroll
  for (int i = 0; i < 4; ++i) o[i] = zeroV;
  v8f o5 = zeroV;                            // running sum-of-exp accumulator (all columns equal)
  float runM[8];
#pragma unroll
  for (int r = 0; r < 8; ++r) runM[r] = -3.0e38f;

  const _Float16* q2g = q2h + (size_t)b * NTOK * DIMD;
  for (int it = 0; it < NTOK / 32; ++it) {
    if (wave == 0)                                      // one TDM issue per tile
      tdm_load_tile(q2g + (size_t)it * 32 * DIMD, sq2[it & 1]);
    __syncthreads();
    const _Float16* tile = sq2[it & 1];

    v8f s[2];
#pragma unroll
    for (int t = 0; t < 2; ++t) {
      HOp b0 = load_B(tile, DIMD, t * 16 + col, 0, lane);
      HOp b1 = load_B(tile, DIMD, t * 16 + col, 32, lane);
      v8f st = wmma16(a0, b0, zeroV);
      st = wmma16(a1, b1, st);
#pragma unroll
      for (int r = 0; r < 8; ++r) st[r] *= 0.125f;       // 1/sqrt(64)
      s[t] = st;
    }
    // rowwise chunk max (reduce across 16 lanes of each half)
    float cm[8];
#pragma unroll
    for (int r = 0; r < 8; ++r) cm[r] = fmaxf(s[0][r], s[1][r]);
#pragma unroll
    for (int m = 1; m <= 8; m <<= 1)
#pragma unroll
      for (int r = 0; r < 8; ++r) cm[r] = fmaxf(cm[r], __shfl_xor(cm[r], m, 32));
    float alpha[8];
#pragma unroll
    for (int r = 0; r < 8; ++r) {
      float nm = fmaxf(runM[r], cm[r]);
      alpha[r] = __expf(runM[r] - nm);
      runM[r] = nm;
    }
#pragma unroll
    for (int t = 0; t < 2; ++t)
#pragma unroll
      for (int r = 0; r < 8; ++r)
        s[t][r] = __expf(s[t][r] - runM[r]);
    // rescale all accumulators (incl. the sum column) by alpha
#pragma unroll
    for (int dt = 0; dt < 4; ++dt)
#pragma unroll
      for (int r = 0; r < 8; ++r) o[dt][r] *= alpha[r];
#pragma unroll
    for (int r = 0; r < 8; ++r) o5[r] *= alpha[r];

    // convert P (C-layout f32) -> A-layout f16 via wave-private LDS patch
    _Float16* mysp = sp[wave];
#pragma unroll
    for (int t = 0; t < 2; ++t)
#pragma unroll
      for (int r = 0; r < 8; ++r)
        mysp[(rb + r) * 32 + t * 16 + col] = (_Float16)s[t][r];
    HOp pa = load_A(mysp, 32, col, 0, lane);
#pragma unroll
    for (int dt = 0; dt < 4; ++dt) {
      HOp bv = load_BT(tile, DIMD, dt * 16 + col, lane);  // q2 chunk as (K=m x N=d)
      o[dt] = wmma16(pa, bv, o[dt]);
    }
    o5 = wmma16(pa, onesB, o5);                           // row-sum of P, no shuffles needed
  }
  float invL[8];
#pragma unroll
  for (int r = 0; r < 8; ++r) invL[r] = 1.0f / o5[r];
  float* v2base = v2 + (size_t)(b * NTOK + nrow0) * DIMD;
#pragma unroll
  for (int dt = 0; dt < 4; ++dt)
#pragma unroll
    for (int r = 0; r < 8; ++r)
      v2base[(rb + r) * DIMD + dt * 16 + col] = o[dt][r] * invL[r];
  if (col == 0) {
#pragma unroll
    for (int r = 0; r < 8; ++r) {
      Mst[b * NTOK + nrow0 + rb + r] = runM[r];
      Lst[b * NTOK + nrow0 + rb + r] = o5[r];
    }
  }
}

// ------- Pass B: v1(b,m,d) = sum_n exp(s(n,m)-M_n)/L_n * q1(n,d); store transposed -------
__global__ __launch_bounds__(256) void attn_passB(const _Float16* __restrict__ q1h,
                                                  const _Float16* __restrict__ q2h,
                                                  const float* __restrict__ Mst,
                                                  const float* __restrict__ Lst,
                                                  float* __restrict__ v1img) {
  __shared__ alignas(16) _Float16 sq1[2][32 * DIMD];
  __shared__ alignas(16) _Float16 sp[8][16 * 32];
  const int tid = threadIdx.x, lane = tid & 31, wave = tid >> 5;
  const int col = lane & 15, rb = (lane & 16) ? 8 : 0;
  const int b = blockIdx.x >> 5;
  const int mrow0 = ((blockIdx.x & 31) << 7) + wave * 16;

  const _Float16* q2base = q2h + (size_t)(b * NTOK + mrow0) * DIMD;
  HOp a0 = load_A(q2base, DIMD, col, 0, lane);
  HOp a1 = load_A(q2base, DIMD, col, 32, lane);
  v8f zeroV = {};
  v8f o[4];
#pragma unroll
  for (int i = 0; i < 4; ++i) o[i] = zeroV;

  const _Float16* q1g = q1h + (size_t)b * NTOK * DIMD;
  const float* Mb = Mst + b * NTOK;
  const float* Lb = Lst + b * NTOK;
  for (int it = 0; it < NTOK / 32; ++it) {
    if (wave == 0)
      tdm_load_tile(q1g + (size_t)it * 32 * DIMD, sq1[it & 1]);
    __syncthreads();
    const _Float16* tile = sq1[it & 1];
    _Float16* mysp = sp[wave];
#pragma unroll
    for (int t = 0; t < 2; ++t) {
      HOp b0 = load_B(tile, DIMD, t * 16 + col, 0, lane);
      HOp b1 = load_B(tile, DIMD, t * 16 + col, 32, lane);
      v8f st = wmma16(a0, b0, zeroV);
      st = wmma16(a1, b1, st);
      float Mn = Mb[it * 32 + t * 16 + col];
      float Li = 1.0f / Lb[it * 32 + t * 16 + col];
#pragma unroll
      for (int r = 0; r < 8; ++r) {
        float p = __expf(st[r] * 0.125f - Mn) * Li;
        mysp[(rb + r) * 32 + t * 16 + col] = (_Float16)p;
      }
    }
    HOp pa = load_A(mysp, 32, col, 0, lane);
#pragma unroll
    for (int dt = 0; dt < 4; ++dt) {
      HOp bv = load_BT(tile, DIMD, dt * 16 + col, lane);  // q1 chunk as (K=n x N=d)
      o[dt] = wmma16(pa, bv, o[dt]);
    }
  }
  // store transposed: v1img[b][d][m]
  float* base = v1img + (size_t)b * DIMD * NTOK;
#pragma unroll
  for (int dt = 0; dt < 4; ++dt)
#pragma unroll
    for (int r = 0; r < 8; ++r)
      base[(size_t)(dt * 16 + col) * NTOK + mrow0 + rb + r] = o[dt][r];
}

extern "C" void kernel_launch(void* const* d_in, const int* in_sizes, int n_in,
                              void* d_out, int out_size, void* d_ws, size_t ws_size,
                              hipStream_t stream) {
  (void)in_sizes; (void)n_in; (void)out_size; (void)ws_size;
  const float* x  = (const float*)d_in[0];
  const float* y  = (const float*)d_in[1];
  const float* w1 = (const float*)d_in[2];
  const float* b1 = (const float*)d_in[3];
  const float* w2 = (const float*)d_in[4];
  const float* b2 = (const float*)d_in[5];

  char* ws = (char*)d_ws;
  _Float16* q1h = (_Float16*)ws;                       // 2 MB
  _Float16* q2h = (_Float16*)(ws + (1u << 21));        // 2 MB
  float* Mst = (float*)(ws + (1u << 22));              // 64 KB
  float* Lst = (float*)(ws + (1u << 22) + (1u << 16)); // 64 KB

  float* v2    = (float*)d_out;                               // (B, N1, D)
  float* v1img = (float*)d_out + (size_t)NB * NTOK * DIMD;    // (B, D, H, W)

  prep_q1<<<4096, 256, 0, stream>>>(x, w1, b1, q1h);
  prep_q2<<<4096, 256, 0, stream>>>(y, w2, b2, q2h);
  attn_passA<<<NB * 32, 256, 0, stream>>>(q1h, q2h, v2, Mst, Lst);
  attn_passB<<<NB * 32, 256, 0, stream>>>(q1h, q2h, Mst, Lst, v1img);
}
